// RWKVTimeMixer_23055384445023
// MI455X (gfx1250) — matrix operand order repaired and checked by reading, not compiled, not run on hardware
//
#include <hip/hip_runtime.h>
#include <math.h>

// ---------------------------------------------------------------------------
// Types for CDNA5 WMMA (wave32).
// ---------------------------------------------------------------------------
typedef __attribute__((ext_vector_type(16))) __bf16 v16bf;
typedef __attribute__((ext_vector_type(8)))  __bf16 v8bf;
typedef __attribute__((ext_vector_type(8)))  float  v8f;

static __device__ __forceinline__ float sigf(float x) {
    return 1.0f / (1.0f + __builtin_expf(-x));
}

// Async global->LDS copy of 16B per lane (CDNA5, ASYNCcnt-tracked).
// The instruction offset is added to both the LDS and the global address.
static __device__ __forceinline__ void async_ld_lds_b128(unsigned ldsOff,
                                                         const void* gaddr) {
    asm volatile("global_load_async_to_lds_b128 %0, %1, off"
                 :: "v"(ldsOff), "v"(gaddr) : "memory");
}
static __device__ __forceinline__ void async_ld_lds_b128_o16(unsigned ldsOff,
                                                             const void* gaddr) {
    asm volatile("global_load_async_to_lds_b128 %0, %1, off offset:16"
                 :: "v"(ldsOff), "v"(gaddr) : "memory");
}
static __device__ __forceinline__ void wait_async0() {
    asm volatile("s_wait_asynccnt 0x0" ::: "memory");
}

// ---------------------------------------------------------------------------
// fp32 -> bf16 convert (for weights)
// ---------------------------------------------------------------------------
__global__ __launch_bounds__(256)
void cvt_f32_bf16(const float* __restrict__ in, __bf16* __restrict__ out, int n) {
    int i = blockIdx.x * blockDim.x + threadIdx.x;
    if (i < n) out[i] = (__bf16)in[i];
}

// ---------------------------------------------------------------------------
// Fused LayerNorm (over C) + cast to bf16.  One 256-thread block per row.
// ---------------------------------------------------------------------------
__global__ __launch_bounds__(256)
void ln_bf16(const float* __restrict__ x, const float* __restrict__ gamma,
             const float* __restrict__ beta, __bf16* __restrict__ xn, int C) {
    const int row = blockIdx.x;
    const float* xr = x + (size_t)row * C;
    __bf16* outr = xn + (size_t)row * C;

    float s1 = 0.f, s2 = 0.f;
    for (int i = threadIdx.x; i < C; i += 256) {
        float v = xr[i];
        s1 += v;
        s2 += v * v;
    }
    #pragma unroll
    for (int off = 16; off > 0; off >>= 1) {
        s1 += __shfl_down(s1, off);
        s2 += __shfl_down(s2, off);
    }
    __shared__ float red1[8], red2[8];
    const int wave = threadIdx.x >> 5;
    const int lane = threadIdx.x & 31;
    if (lane == 0) { red1[wave] = s1; red2[wave] = s2; }
    __syncthreads();
    float t1 = 0.f, t2 = 0.f;
    #pragma unroll
    for (int w = 0; w < 8; ++w) { t1 += red1[w]; t2 += red2[w]; }

    const float invC = 1.0f / (float)C;
    const float mu = t1 * invC;
    const float var = t2 * invC - mu * mu;
    const float rstd = rsqrtf(var + 1e-5f);

    for (int i = threadIdx.x; i < C; i += 256) {
        float v = (xr[i] - mu) * rstd * gamma[i] + beta[i];
        outr[i] = (__bf16)v;
    }
}

// ---------------------------------------------------------------------------
// bf16 WMMA GEMM:  Y[m,n] = sum_k A[m,k] * W[n,k]   (A: MxK, W: NxK row-major)
//
// Block: 256 threads = 8 waves; block tile 256(M) x 64(N).
// Wave tile: 32(M) x 64(N) = 8 WMMA accumulators (2 M-frags x 4 N-frags).
// B tile (64n x 64k bf16 = 8KB) is staged in LDS via async global->LDS loads,
// double-buffered: one barrier per 64-K stage.  A fragments load directly.
// Per 32-K step: batch-load all 4 B fragments (8x ds_load_b128), then issue
// the 8 WMMAs back-to-back so LDS latency overlaps the matrix pipe.
// ---------------------------------------------------------------------------
__global__ __launch_bounds__(256)
void gemm_bf16_wmma(const __bf16* __restrict__ A, const __bf16* __restrict__ W,
                    float* __restrict__ Y, int M, int N, int K) {
    __shared__ __bf16 Bsm[2][64 * 64];          // [buf][n*64 + k], 8KB each

    const int tid   = threadIdx.x;
    const int lane  = tid & 31;
    const int wave  = tid >> 5;
    const int l15   = lane & 15;
    const int lhalf = lane >> 4;                // 0: lanes 0-15, 1: lanes 16-31

    const int m0 = blockIdx.y * 256 + wave * 32;
    const int n0 = blockIdx.x * 64;
    if (m0 >= M || n0 >= N) return;

    // --- async B-stage assignment: each thread copies 32B (16 bf16) ---------
    const int fn = tid >> 2;                     // 0..63 : n within tile
    const int fk = (tid & 3) * 16;               // 0,16,32,48 : k within stage
    const __bf16* wsrc = W + (size_t)(n0 + fn) * K + fk;     // + kb per stage
    const unsigned ldsRow = (unsigned)(fn * 64 + fk) * 2u;   // byte offset in tile

    const unsigned ldsBuf0 = (unsigned)(size_t)(&Bsm[0][0]);
    const unsigned ldsBuf1 = (unsigned)(size_t)(&Bsm[1][0]);

    const __bf16* Arow0 = A + (size_t)(m0 + l15) * K;
    const __bf16* Arow1 = A + (size_t)(m0 + 16 + l15) * K;

    v8f acc[2][4];
    #pragma unroll
    for (int mi = 0; mi < 2; ++mi)
        #pragma unroll
        for (int j = 0; j < 4; ++j) acc[mi][j] = (v8f){};

    const int nstage = K / 64;

    // prologue: fill buffer 0
    async_ld_lds_b128(ldsBuf0 + ldsRow, wsrc);
    async_ld_lds_b128_o16(ldsBuf0 + ldsRow, wsrc);
    wait_async0();
    __syncthreads();

    for (int s = 0; s < nstage; ++s) {
        const int kb = s * 64;
        const __bf16* Bcur = &Bsm[s & 1][0];

        // prefetch next stage into the other buffer (safe: its readers
        // finished at the barrier that ended iteration s-1)
        if (s + 1 < nstage) {
            const unsigned dst = ((s + 1) & 1) ? ldsBuf1 : ldsBuf0;
            const __bf16* src = wsrc + (kb + 64);
            async_ld_lds_b128(dst + ldsRow, src);
            async_ld_lds_b128_o16(dst + ldsRow, src);
        }

        #pragma unroll
        for (int kk = 0; kk < 64; kk += 32) {
            // A fragments (16x32 bf16): lanes 0-15 hold K {0..15},
            // lanes 16-31 hold K {16..31} of the 32-wide step.
            const int abase = kb + kk + (lhalf << 4);
            v16bf afrag[2];
            #pragma unroll
            for (int mi = 0; mi < 2; ++mi) {
                const __bf16* ar = mi ? Arow1 : Arow0;
                v8bf alo = *(const v8bf*)(ar + abase);
                v8bf ahi = *(const v8bf*)(ar + abase + 8);
                #pragma unroll
                for (int i = 0; i < 8; ++i) {
                    afrag[mi][i] = alo[i];
                    afrag[mi][i + 8] = ahi[i];
                }
            }

            // Batch-load all B fragments (32x16 bf16 each) from LDS first.
            v16bf bfr[4];
            #pragma unroll
            for (int j = 0; j < 4; ++j) {
                const __bf16* bp = Bcur + (size_t)(j * 16 + l15) * 64
                                        + kk + (lhalf << 4);
                bfr[j] = *(const v16bf*)bp;
            }

            // WMMA burst: 8 back-to-back matrix ops.
            #pragma unroll
            for (int j = 0; j < 4; ++j) {
                acc[0][j] = __builtin_amdgcn_wmma_f32_16x16x32_bf16(
                    false, afrag[0], false, bfr[j], (short)0, acc[0][j], false, false);
                acc[1][j] = __builtin_amdgcn_wmma_f32_16x16x32_bf16(
                    false, afrag[1], false, bfr[j], (short)0, acc[1][j], false, false);
            }
        }

        wait_async0();        // this wave's prefetch landed
        __syncthreads();      // everyone's prefetch landed; reads of Bcur done
    }

    // C/D layout: VGPR r, lanes 0-15 -> (M=r, N=lane); lanes 16-31 -> (M=r+8).
    #pragma unroll
    for (int mi = 0; mi < 2; ++mi) {
        const int rbase = m0 + mi * 16 + 8 * lhalf;
        #pragma unroll
        for (int vi = 0; vi < 8; ++vi) {
            float* yr = Y + (size_t)(rbase + vi) * N + n0 + l15;
            yr[0]  = acc[mi][0][vi];
            yr[16] = acc[mi][1][vi];
            yr[32] = acc[mi][2][vi];
            yr[48] = acc[mi][3][vi];
        }
    }
}

// ---------------------------------------------------------------------------
// Chunked time-scan.  wkv[t] = tf*kv[t] + td * sum_{s<t} kv[s]*td^s
// ---------------------------------------------------------------------------
__global__ __launch_bounds__(256)
void scan_partial(const float* __restrict__ k, const float* __restrict__ v,
                  const float* __restrict__ time_decay, float* __restrict__ csum,
                  int B, int T, int C, int nch) {
    const int idx = blockIdx.x * blockDim.x + threadIdx.x;
    const int total = B * nch * C;
    if (idx >= total) return;
    const int c = idx % C;
    const int chunk = (idx / C) % nch;
    const int b = idx / (C * nch);

    const float td = sigf(time_decay[c]);
    const int tch = T / nch;
    const int t0 = chunk * tch;
    float p = __builtin_exp2f((float)t0 * __builtin_log2f(td));
    float s = 0.f;
    size_t off = ((size_t)b * T + t0) * C + c;
    for (int i = 0; i < tch; ++i) {
        s += k[off] * v[off] * p;
        p *= td;
        off += C;
    }
    csum[((size_t)b * nch + chunk) * C + c] = s;
}

__global__ __launch_bounds__(256)
void scan_prefix(float* __restrict__ csum, int B, int C, int nch) {
    const int idx = blockIdx.x * blockDim.x + threadIdx.x;
    if (idx >= B * C) return;
    const int c = idx % C;
    const int b = idx / C;
    float run = 0.f;
    for (int j = 0; j < nch; ++j) {
        size_t o = ((size_t)b * nch + j) * C + c;
        float s = csum[o];
        csum[o] = run;   // exclusive prefix
        run += s;
    }
}

__global__ __launch_bounds__(256)
void scan_emit(const float* __restrict__ k, const float* __restrict__ v,
               const float* __restrict__ r, const float* __restrict__ csum,
               const float* __restrict__ time_decay,
               const float* __restrict__ time_first,
               __bf16* __restrict__ y2, int B, int T, int C, int nch) {
    const int idx = blockIdx.x * blockDim.x + threadIdx.x;
    const int total = B * nch * C;
    if (idx >= total) return;
    const int c = idx % C;
    const int chunk = (idx / C) % nch;
    const int b = idx / (C * nch);

    const float td = sigf(time_decay[c]);
    const float tf = sigf(time_first[c]);
    const int tch = T / nch;
    const int t0 = chunk * tch;
    float p = __builtin_exp2f((float)t0 * __builtin_log2f(td));
    float S = csum[((size_t)b * nch + chunk) * C + c];  // prefix up to t0-1
    size_t off = ((size_t)b * T + t0) * C + c;
    for (int i = 0; i < tch; ++i) {
        float kv = k[off] * v[off];
        float wkv = tf * kv + td * S;   // S holds csum[t-1]
        S += kv * p;
        p *= td;
        y2[off] = (__bf16)(sigf(r[off]) * wkv);
        off += C;
    }
}

// ---------------------------------------------------------------------------
// Launcher
// ---------------------------------------------------------------------------
extern "C" void kernel_launch(void* const* d_in, const int* in_sizes, int n_in,
                              void* d_out, int out_size, void* d_ws, size_t ws_size,
                              hipStream_t stream) {
    const float* x          = (const float*)d_in[0];
    const float* time_decay = (const float*)d_in[1];
    const float* time_first = (const float*)d_in[2];
    const float* Wk         = (const float*)d_in[3];
    const float* Wv         = (const float*)d_in[4];
    const float* Wr         = (const float*)d_in[5];
    const float* Wo         = (const float*)d_in[6];
    const float* ln_gamma   = (const float*)d_in[7];
    const float* ln_beta    = (const float*)d_in[8];

    const int C = in_sizes[1];           // 1024
    const int M = in_sizes[0] / C;       // B*T = 16384
    const int T = 4096;
    const int B = M / T;                 // 4
    const int NCH = 64;                  // scan chunks along T

    // Workspace layout
    char* w = (char*)d_ws;
    __bf16* xn_bf  = (__bf16*)w;                    w += (size_t)M * C * sizeof(__bf16);
    __bf16* y2_bf  = (__bf16*)w;                    w += (size_t)M * C * sizeof(__bf16);
    __bf16* Wk_bf  = (__bf16*)w;                    w += (size_t)C * C * sizeof(__bf16);
    __bf16* Wv_bf  = (__bf16*)w;                    w += (size_t)C * C * sizeof(__bf16);
    __bf16* Wr_bf  = (__bf16*)w;                    w += (size_t)C * C * sizeof(__bf16);
    __bf16* Wo_bf  = (__bf16*)w;                    w += (size_t)C * C * sizeof(__bf16);
    float*  kbuf   = (float*)w;                     w += (size_t)M * C * sizeof(float);
    float*  vbuf   = (float*)w;                     w += (size_t)M * C * sizeof(float);
    float*  csum   = (float*)w;                     w += (size_t)B * NCH * C * sizeof(float);
    float*  rbuf   = (float*)d_out;                 // r parked in d_out, overwritten later
    float*  out    = (float*)d_out;

    // 1) weights -> bf16
    {
        int n = C * C;
        dim3 g((n + 255) / 256), blk(256);
        cvt_f32_bf16<<<g, blk, 0, stream>>>(Wk, Wk_bf, n);
        cvt_f32_bf16<<<g, blk, 0, stream>>>(Wv, Wv_bf, n);
        cvt_f32_bf16<<<g, blk, 0, stream>>>(Wr, Wr_bf, n);
        cvt_f32_bf16<<<g, blk, 0, stream>>>(Wo, Wo_bf, n);
    }

    // 2) LayerNorm + bf16 cast
    ln_bf16<<<dim3(M), dim3(256), 0, stream>>>(x, ln_gamma, ln_beta, xn_bf, C);

    // 3) projections k, v, r (WMMA bf16, f32 accumulate)
    dim3 ggrid((C + 63) / 64, (M + 255) / 256), gblk(256);
    gemm_bf16_wmma<<<ggrid, gblk, 0, stream>>>(xn_bf, Wk_bf, kbuf, M, C, C);
    gemm_bf16_wmma<<<ggrid, gblk, 0, stream>>>(xn_bf, Wv_bf, vbuf, M, C, C);
    gemm_bf16_wmma<<<ggrid, gblk, 0, stream>>>(xn_bf, Wr_bf, rbuf, M, C, C);

    // 4) chunked scan
    {
        int total = B * NCH * C;
        dim3 g((total + 255) / 256), blk(256);
        scan_partial<<<g, blk, 0, stream>>>(kbuf, vbuf, time_decay, csum, B, T, C, NCH);
        int tot2 = B * C;
        scan_prefix<<<dim3((tot2 + 255) / 256), blk, 0, stream>>>(csum, B, C, NCH);
        scan_emit<<<g, blk, 0, stream>>>(kbuf, vbuf, rbuf, csum, time_decay,
                                         time_first, y2_bf, B, T, C, NCH);
    }

    // 5) output projection (overwrites d_out)
    gemm_bf16_wmma<<<ggrid, gblk, 0, stream>>>(y2_bf, Wo_bf, out, M, C, C);
}
